// U_Y_encoder_5583457485491
// MI455X (gfx1250) — matrix operand order repaired and checked by reading, not compiled, not run on hardware
//
#include <hip/hip_runtime.h>

// ---------- types ----------
typedef __bf16 v16bf __attribute__((ext_vector_type(16)));
typedef float  v8f   __attribute__((ext_vector_type(8)));
typedef unsigned int u32x4 __attribute__((ext_vector_type(4)));
typedef float  f32x4 __attribute__((ext_vector_type(4)));

#define GCN_H 128   // hidden width == F_IN+1 == Wct width
#define GCN_Z 64

// ---------- degree / norm ----------
__global__ __launch_bounds__(256) void k_init_deg(float* deg, int n) {
    int i = blockIdx.x * 256 + threadIdx.x;
    if (i < n) deg[i] = 1.0f;               // self-loop
}
__global__ __launch_bounds__(256) void k_acc_deg(const int* __restrict__ dst, float* deg, int e) {
    int i = blockIdx.x * 256 + threadIdx.x;
    if (i < e) atomicAdd(&deg[dst[i]], 1.0f);
}
__global__ __launch_bounds__(256) void k_rsqrt(float* d, int n) {
    int i = blockIdx.x * 256 + threadIdx.x;
    if (i < n) d[i] = rsqrtf(fmaxf(d[i], 1e-12f));
}

// ---------- h0 = bf16(|cat(x, Y)|) ----------
__global__ __launch_bounds__(256) void k_build_h0(const float* __restrict__ x,
                                                  const float* __restrict__ Y,
                                                  __bf16* __restrict__ h0, int n) {
    int g = blockIdx.x * 256 + threadIdx.x;
    if (g >= n * GCN_H) return;
    int i = g >> 7, c = g & 127;
    float v = (c < 127) ? x[(size_t)i * 127 + c] : Y[i];
    h0[g] = (__bf16)fabsf(v);
}

// ---------- weights: W1t[h][k]=W1[k][h]; Wct[n][k]= n<64 ? Wmu[k][n] : Wls[k][n-64] ----------
__global__ __launch_bounds__(256) void k_prep_w(const float* __restrict__ W1,
                                                const float* __restrict__ Wmu,
                                                const float* __restrict__ Wls,
                                                __bf16* __restrict__ W1t,
                                                __bf16* __restrict__ Wct) {
    int g = blockIdx.x * 256 + threadIdx.x;        // 32768 threads
    if (g < 16384) {
        int nn = g >> 7, k = g & 127;
        W1t[g] = (__bf16)W1[(size_t)k * 128 + nn];
    } else {
        int j = g - 16384;
        int nn = j >> 7, k = j & 127;
        float w = (nn < GCN_Z) ? Wmu[(size_t)k * GCN_Z + nn]
                               : Wls[(size_t)k * GCN_Z + (nn - GCN_Z)];
        Wct[j] = (__bf16)w;
    }
}

// ---------- WMMA GEMM: C[N,128] = A[N,128](bf16) x Bt[128,128](bf16, row n = column n of B)
// Bt staged in LDS once per block; all 4 A fragments hoisted to registers. ----------
__global__ __launch_bounds__(256) void k_gemm_wmma(const __bf16* __restrict__ A,
                                                   const __bf16* __restrict__ Bt,
                                                   float* __restrict__ C,
                                                   int numRowTiles) {
    __shared__ __bf16 Bs[128 * 128];               // 32 KB of the WGP's 320 KB LDS

    // cooperative stage of Bt: 2048 x 16B = 32 KB with 256 threads
#pragma unroll
    for (int it = threadIdx.x; it < 2048; it += 256)
        ((u32x4*)Bs)[it] = ((const u32x4*)Bt)[it];
    __syncthreads();

    const int lane = threadIdx.x & 31;
    const int wave = threadIdx.x >> 5;
    const int tile = blockIdx.x * 8 + wave;
    if (tile >= numRowTiles) return;
    const int row0 = tile * 16;
    const int m  = lane & 15;
    const int hi = lane >> 4;

    // ---- hoist all A fragments (K = 0..127) into registers: 4 frags x 32B/lane ----
    union { u32x4 q[2]; v16bf v; } a[4];
    const __bf16* arow = A + (size_t)(row0 + m) * 128;
#pragma unroll
    for (int ks = 0; ks < 4; ++ks) {
        const __bf16* p = arow + ks * 32 + hi * 8;
        a[ks].q[0] = *(const u32x4*)(p);
        a[ks].q[1] = *(const u32x4*)(p + 16);
    }

    v8f acc[8];
#pragma unroll
    for (int t = 0; t < 8; ++t)
#pragma unroll
        for (int r = 0; r < 8; ++r) acc[t][r] = 0.0f;

#pragma unroll
    for (int ks = 0; ks < 4; ++ks) {
        const int k0 = ks * 32;
#pragma unroll
        for (int t = 0; t < 8; ++t) {
            // B fragment (32x16 bf16): lane n=lane&15 holds K k0+hi*8+[0..7] and k0+16+hi*8+[0..7] of Bs row n
            union { u32x4 q[2]; v16bf v; } b;
            const __bf16* brow = Bs + (size_t)(t * 16 + m) * 128 + k0 + hi * 8;
            b.q[0] = *(const u32x4*)(brow);
            b.q[1] = *(const u32x4*)(brow + 16);
            acc[t] = __builtin_amdgcn_wmma_f32_16x16x32_bf16(
                false, a[ks].v, false, b.v, (short)0, acc[t], false, false);
        }
    }

    // D layout: VGPR r, lanes 0-15 -> M=r, lanes 16-31 -> M=8+r; N=lane&15
#pragma unroll
    for (int t = 0; t < 8; ++t) {
        int n = t * 16 + m;
#pragma unroll
        for (int r = 0; r < 8; ++r)
            C[(size_t)(row0 + r + hi * 8) * 128 + n] = acc[t][r];
    }
}

// ---------- agg = dinv[i]^2 * tmp (self-loop term, no atomics) ----------
__global__ __launch_bounds__(256) void k_selfloop(const float* __restrict__ dinv,
                                                  const float* __restrict__ tmp,
                                                  float* __restrict__ agg, int n) {
    int g = blockIdx.x * 256 + threadIdx.x;        // n*32 vec4 groups
    if (g >= n * 32) return;
    int i = g >> 5;
    float s = dinv[i]; s *= s;
    f32x4 v = ((const f32x4*)tmp)[g];
    v.x *= s; v.y *= s; v.z *= s; v.w *= s;
    ((f32x4*)agg)[g] = v;
}

// ---------- edge scatter: one wave per edge, 128 columns ----------
__global__ __launch_bounds__(256) void k_scatter(const int* __restrict__ src,
                                                 const int* __restrict__ dst,
                                                 const float* __restrict__ dinv,
                                                 const float* __restrict__ tmp,
                                                 float* __restrict__ agg, int e) {
    int g = blockIdx.x * 256 + threadIdx.x;
    int ed = g >> 5;
    int lane = g & 31;
    if (ed >= e) return;
    int s = src[ed], d = dst[ed];
    float w = dinv[s] * dinv[d];
    const float* ts = tmp + (size_t)s * 128;
    float* ad = agg + (size_t)d * 128;
#pragma unroll
    for (int c = lane; c < 128; c += 32)
        atomicAdd(&ad[c], w * ts[c]);
}

// ---------- h = bf16(relu(agg + b1)) ----------
__global__ __launch_bounds__(256) void k_relu_fin(const float* __restrict__ agg,
                                                  const float* __restrict__ b1,
                                                  __bf16* __restrict__ hb, int n) {
    int g = blockIdx.x * 256 + threadIdx.x;
    if (g >= n * GCN_H) return;
    int c = g & 127;
    hb[g] = (__bf16)fmaxf(agg[g] + b1[c], 0.0f);
}

// ---------- finalize: mu = agg[:, :64]+bmu ; logstd = softplus(agg[:, 64:]+bls) ----------
__global__ __launch_bounds__(256) void k_finalize(const float* __restrict__ agg,
                                                  const float* __restrict__ bmu,
                                                  const float* __restrict__ bls,
                                                  float* __restrict__ out, int n) {
    int g = blockIdx.x * 256 + threadIdx.x;
    if (g >= n * GCN_H) return;
    int i = g >> 7, c = g & 127;
    float v = agg[g];
    if (c < GCN_Z) {
        out[(size_t)i * GCN_Z + c] = v + bmu[c];
    } else {
        float u = v + bls[c - GCN_Z];
        // stable softplus: max(u,0) + log1p(exp(-|u|))
        float sp = fmaxf(u, 0.0f) + log1pf(expf(-fabsf(u)));
        out[(size_t)n * GCN_Z + (size_t)i * GCN_Z + (c - GCN_Z)] = sp;
    }
}

static inline size_t alignup(size_t v) { return (v + 255) & ~(size_t)255; }

extern "C" void kernel_launch(void* const* d_in, const int* in_sizes, int n_in,
                              void* d_out, int out_size, void* d_ws, size_t ws_size,
                              hipStream_t stream) {
    const float* x   = (const float*)d_in[0];
    const float* Y   = (const float*)d_in[1];
    const int*   ei  = (const int*)d_in[2];
    const float* W1  = (const float*)d_in[3];
    const float* b1  = (const float*)d_in[4];
    const float* Wmu = (const float*)d_in[5];
    const float* bmu = (const float*)d_in[6];
    const float* Wls = (const float*)d_in[7];
    const float* bls = (const float*)d_in[8];
    (void)n_in; (void)out_size; (void)ws_size;

    const int N = in_sizes[1];          // 100000
    const int E = in_sizes[2] / 2;      // 800000
    const int* src = ei;
    const int* dst = ei + E;

    // workspace carve-out
    char* ws = (char*)d_ws;
    float* dinv  = (float*)ws;  ws += alignup((size_t)N * 4);
    __bf16* hbuf = (__bf16*)ws; ws += alignup((size_t)N * GCN_H * 2);  // h0, reused as h
    __bf16* W1t  = (__bf16*)ws; ws += alignup(16384 * 2);
    __bf16* Wct  = (__bf16*)ws; ws += alignup(16384 * 2);
    float* tmp   = (float*)ws;  ws += alignup((size_t)N * GCN_H * 4);
    float* agg   = (float*)ws;  ws += alignup((size_t)N * GCN_H * 4);

    const int rowTiles = (N + 15) / 16;                 // 6250
    const int gemmBlocks = (rowTiles + 7) / 8;          // 782
    const int nh = N * GCN_H;

    // 1) symmetric normalization coefficients
    k_init_deg<<<(N + 255) / 256, 256, 0, stream>>>(dinv, N);
    k_acc_deg<<<(E + 255) / 256, 256, 0, stream>>>(dst, dinv, E);
    k_rsqrt<<<(N + 255) / 256, 256, 0, stream>>>(dinv, N);

    // 2) inputs / weights to bf16
    k_build_h0<<<(nh + 255) / 256, 256, 0, stream>>>(x, Y, hbuf, N);
    k_prep_w<<<32768 / 256, 256, 0, stream>>>(W1, Wmu, Wls, W1t, Wct);

    // 3) layer 1: tmp = h0 @ W1 ; aggregate ; h = relu(. + b1)
    k_gemm_wmma<<<gemmBlocks, 256, 0, stream>>>(hbuf, W1t, tmp, rowTiles);
    k_selfloop<<<(N * 32 + 255) / 256, 256, 0, stream>>>(dinv, tmp, agg, N);
    k_scatter<<<(E * 32 + 255) / 256, 256, 0, stream>>>(src, dst, dinv, tmp, agg, E);
    k_relu_fin<<<(nh + 255) / 256, 256, 0, stream>>>(agg, b1, hbuf, N);

    // 4) layer 2 (mu || logstd fused): tmp = h @ [Wmu|Wls] ; aggregate ; finalize
    k_gemm_wmma<<<gemmBlocks, 256, 0, stream>>>(hbuf, Wct, tmp, rowTiles);
    k_selfloop<<<(N * 32 + 255) / 256, 256, 0, stream>>>(dinv, tmp, agg, N);
    k_scatter<<<(E * 32 + 255) / 256, 256, 0, stream>>>(src, dst, dinv, tmp, agg, E);
    k_finalize<<<(nh + 255) / 256, 256, 0, stream>>>(agg, bmu, bls, (float*)d_out, N);
}
